// DualCodebookVectorQuantizer_43885975831036
// MI455X (gfx1250) — compile-verified, matrix operand order checked by reading, not run on hardware
//
#include <hip/hip_runtime.h>
#include <stdint.h>

// ---------------- problem constants ----------------
#define RVQ_DEPTH 6
constexpr int EDIM   = 512;              // 128 * 2 * 2
constexpr int M_ROWS = 31752;            // 8 * 63 * 63
constexpr int M_PAD  = 31872;            // 249 * 128 (multiple of 128-row tile)
constexpr int NE_S   = 1024;
constexpr int NE_T   = 512;
constexpr int IMG_N  = 8 * 128 * 64 * 64; // 4194304
constexpr float EPSF = 1e-8f;

// ---------------- CDNA5 WMMA types ----------------
typedef __bf16 bf16_t;
typedef bf16_t bf16x16 __attribute__((ext_vector_type(16)));
typedef float  f32x8   __attribute__((ext_vector_type(8)));
typedef unsigned int u32x4 __attribute__((ext_vector_type(4)));

union BFrag { u32x4 q[2]; bf16x16 v; };

__device__ __forceinline__ unsigned short f2bf(float f) {
  unsigned int u = __float_as_uint(f);
  u += 0x7FFFu + ((u >> 16) & 1u);      // round-to-nearest-even
  return (unsigned short)(u >> 16);
}

// ---------------- unfold: z -> residual (f32 + bf16) + row norms ----------------
__global__ __launch_bounds__(256) void k_unfold(const float* __restrict__ z,
    float* __restrict__ r, unsigned short* __restrict__ rbf,
    float* __restrict__ invRow)
{
  __shared__ float sred[256];
  const int row = blockIdx.x;
  float s = 0.f;
  if (row < M_ROWS) {
    const int j = row % 63;
    const int t = row / 63;
    const int i = t % 63;
    const int b = t / 63;
    for (int col = threadIdx.x; col < EDIM; col += 256) {
      const int c  = col >> 2;
      const int ki = (col >> 1) & 1;
      const int kj = col & 1;
      const float v = z[(((size_t)b * 128 + c) * 64 + (i + ki)) * 64 + (j + kj)];
      const size_t o = (size_t)row * EDIM + col;
      r[o]   = v;
      rbf[o] = f2bf(v);
      s += v * v;
    }
  } else { // pad rows stay exactly zero so WMMA tiles are harmless
    for (int col = threadIdx.x; col < EDIM; col += 256) {
      const size_t o = (size_t)row * EDIM + col;
      r[o] = 0.f; rbf[o] = 0;
    }
  }
  sred[threadIdx.x] = s; __syncthreads();
  for (int st = 128; st > 0; st >>= 1) {
    if (threadIdx.x < st) sred[threadIdx.x] += sred[threadIdx.x + st];
    __syncthreads();
  }
  if (threadIdx.x == 0) invRow[row] = 1.f / fmaxf(sqrtf(sred[0]), EPSF);
}

// ---------------- codebook prep: bf16 copy + ||cb||^2 + 1/max(||cb||,eps) ----------------
__global__ __launch_bounds__(256) void k_cbprep(const float* __restrict__ cb,
    unsigned short* __restrict__ cbbf, float* __restrict__ cbsq,
    float* __restrict__ invcb)
{
  __shared__ float sred[256];
  const int row = blockIdx.x;
  float s = 0.f;
  for (int col = threadIdx.x; col < EDIM; col += 256) {
    const size_t o = (size_t)row * EDIM + col;
    const float v = cb[o];
    cbbf[o] = f2bf(v);
    s += v * v;
  }
  sred[threadIdx.x] = s; __syncthreads();
  for (int st = 128; st > 0; st >>= 1) {
    if (threadIdx.x < st) sred[threadIdx.x] += sred[threadIdx.x + st];
    __syncthreads();
  }
  if (threadIdx.x == 0) {
    cbsq[row]  = sred[0];
    invcb[row] = 1.f / fmaxf(sqrtf(sred[0]), EPSF);
  }
}

// ---------------- WMMA GEMM skeleton with async-LDS B staging ----------------
// MODE 0: residual-VQ distance argmin  (colAux = cb_sq, writes idxOut)
// MODE 1: fused relu(0.5 - cos) sum    (colAux = inv||cb||, atomicAdd lossAcc)
// MODE 2: store cos matrix             (colAux = inv||cb||, writes cosOut)
// Block = 256 threads = 8 waves; wave owns 16 rows.
// B is staged panel-by-panel (64 cols x 512 K = 64KB, contiguous in global)
// into LDS with global_load_async_to_lds_b128 (ASYNCcnt), 16B-chunk XOR
// swizzle so fragment reads (stride 1024B per lane) are bank-conflict free.
template <int MODE>
__global__ __launch_bounds__(256) void k_gemm(
    const unsigned short* __restrict__ Abf,   // M_PAD x 512 bf16 bits
    const unsigned short* __restrict__ Bbf,   // Ne x 512 bf16 bits
    const float* __restrict__ colAux,
    const float* __restrict__ invRow,
    int*   __restrict__ idxOut,
    float* __restrict__ cosOut,
    float* __restrict__ lossAcc,
    int Ne)
{
  __shared__ unsigned short sB[64 * EDIM];  // 64KB B panel, swizzled

  const int lane   = threadIdx.x & 31;
  const int wave   = threadIdx.x >> 5;
  const int mhalf  = lane >> 4;   // K-half select per ISA 16-bit layout
  const int mrow   = lane & 15;   // A: row index / B: column index
  const int rowbase = blockIdx.x * 128 + wave * 16;
  const unsigned short* aBase = Abf + (size_t)(rowbase + mrow) * EDIM + 8 * mhalf;

  const unsigned ldsBase = (unsigned)(size_t)(void*)&sB[0];
  const u32x4* sBq = (const u32x4*)&sB[0];  // 16B chunk view

  float bestVal[8]; int bestIdx[8];
  #pragma unroll
  for (int v = 0; v < 8; ++v) { bestVal[v] = 3.4e38f; bestIdx[v] = 0; }
  float lossLocal = 0.f;

  const int nPanels = Ne >> 6;                // 64 columns per panel
  for (int pg = 0; pg < nPanels; ++pg) {
    // ---- stage panel: contiguous 64KB global -> LDS via async DMA ----
    {
      const unsigned long long gbase =
          (unsigned long long)(uintptr_t)(Bbf + (size_t)pg * 64 * EDIM);
      #pragma unroll
      for (int i = 0; i < 16; ++i) {
        const int g    = threadIdx.x + i * 256;   // 16B chunk id 0..4095
        const int colP = g >> 6;                  // 64 chunks per codebook row
        const int ch   = g & 63;
        const unsigned dst  = ldsBase +
            ((unsigned)((colP << 6) | (ch ^ (colP & 15))) << 4);
        const unsigned goff = (unsigned)g << 4;
        asm volatile("global_load_async_to_lds_b128 %0, %1, %2"
                     :: "v"(dst), "v"(goff), "s"(gbase) : "memory");
      }
      asm volatile("s_wait_asynccnt 0x0" ::: "memory");
      __syncthreads();
    }

    // ---- compute 4 chunk-columns x 16 ksteps on the panel ----
    f32x8 acc[4] = {};
    for (int ks = 0; ks < EDIM / 32; ++ks) {
      const int k0 = ks * 32;
      BFrag a;
      a.q[0] = *(const u32x4*)(aBase + k0);        // K = k0 + 8*mhalf .. +7
      a.q[1] = *(const u32x4*)(aBase + k0 + 16);   // K = k0 + 16 + 8*mhalf .. +7
      const int ch0 = 4 * ks + mhalf;              // 16B chunk within row
      const int ch1 = ch0 + 2;
      #pragma unroll
      for (int cc = 0; cc < 4; ++cc) {
        const int colP = cc * 16 + mrow;
        const int sw = colP & 15;
        BFrag b;
        b.q[0] = sBq[(colP << 6) | (ch0 ^ sw)];
        b.q[1] = sBq[(colP << 6) | (ch1 ^ sw)];
        acc[cc] = __builtin_amdgcn_wmma_f32_16x16x32_bf16(
            false, a.v, false, b.v, (short)0, acc[cc], false, false);
      }
    }

    #pragma unroll
    for (int cc = 0; cc < 4; ++cc) {
      const int col = pg * 64 + cc * 16 + mrow;
      if (MODE == 0) {
        const float csq = colAux[col];
        #pragma unroll
        for (int v = 0; v < 8; ++v) {
          float val = csq - 2.f * acc[cc][v];  // ||r||^2 constant per row -> drop
          int   idx = col;
          #pragma unroll
          for (int off = 1; off < 16; off <<= 1) {  // min over 16 cols in half-wave
            const float ov = __shfl_xor(val, off, 32);
            const int   oi = __shfl_xor(idx, off, 32);
            if (ov < val || (ov == val && oi < idx)) { val = ov; idx = oi; }
          }
          if (val < bestVal[v]) { bestVal[v] = val; bestIdx[v] = idx; }
        }
      } else {
        const float invc = colAux[col];
        #pragma unroll
        for (int v = 0; v < 8; ++v) {
          const int row = rowbase + v + 8 * mhalf;
          const float cosv = acc[cc][v] * invRow[row] * invc;
          if (MODE == 1) {
            if (row < M_ROWS) {
              const float t = 0.5f - cosv;
              lossLocal += t > 0.f ? t : 0.f;
            }
          } else {
            cosOut[(size_t)row * Ne + col] = cosv;
          }
        }
      }
    }
    __syncthreads();  // protect sB before next panel staging
  }

  if (MODE == 0) {
    if (mrow == 0) {  // lanes 0 / 16 each hold 8 finalized rows
      #pragma unroll
      for (int v = 0; v < 8; ++v) {
        const int row = rowbase + v + 8 * mhalf;
        if (row < M_ROWS) idxOut[row] = bestIdx[v];
      }
    }
  }
  if (MODE == 1) {
    #pragma unroll
    for (int off = 1; off < 32; off <<= 1) lossLocal += __shfl_xor(lossLocal, off, 32);
    if (lane == 0) atomicAdd(lossAcc, lossLocal);
  }
}

// ---------------- residual / zq update + index emit ----------------
__global__ __launch_bounds__(128) void k_update(
    float* __restrict__ r, unsigned short* __restrict__ rbf,
    float* __restrict__ zq, const float* __restrict__ cb,
    const int* __restrict__ idxArr, float* __restrict__ idxOutF, int depth)
{
  const int row = blockIdx.x;           // grid = M_ROWS, pad rows untouched
  const int idx = idxArr[row];
  const float* crow = cb + (size_t)idx * EDIM;
  for (int col = threadIdx.x; col < EDIM; col += 128) {
    const size_t o = (size_t)row * EDIM + col;
    const float d  = crow[col];
    const float nr = r[o] - d;
    r[o]   = nr;
    rbf[o] = f2bf(nr);
    zq[o] += d;
  }
  if (threadIdx.x == 0) idxOutF[(size_t)row * RVQ_DEPTH + depth] = (float)idx;
}

// ---------------- fold with overlap averaging + fused codebook loss ----------------
__global__ __launch_bounds__(256) void k_fold(const float* __restrict__ zqflat,
    const float* __restrict__ z, float* __restrict__ outZq,
    float scale, int first, float* __restrict__ lossAcc)
{
  __shared__ float sred[256];
  const int gid = blockIdx.x * 256 + threadIdx.x;
  float local = 0.f;
  if (gid < IMG_N) {
    const int j = gid & 63, i = (gid >> 6) & 63;
    const int c = (gid >> 12) & 127, b = gid >> 19;
    float sum = 0.f, cnt = 0.f;
    #pragma unroll
    for (int ki = 0; ki < 2; ++ki)
      #pragma unroll
      for (int kj = 0; kj < 2; ++kj) {
        const int pi = i - ki, pj = j - kj;
        if (pi >= 0 && pi < 63 && pj >= 0 && pj < 63) {
          const int row = b * 3969 + pi * 63 + pj;
          sum += zqflat[(size_t)row * EDIM + (c * 4 + ki * 2 + kj)];
          cnt += 1.f;
        }
      }
    const float v = sum / cnt;
    if (first) outZq[gid] = scale * v;
    else       outZq[gid] += scale * v;
    const float dz = v - z[gid];
    local = dz * dz;
  }
  sred[threadIdx.x] = local; __syncthreads();
  for (int st = 128; st > 0; st >>= 1) {
    if (threadIdx.x < st) sred[threadIdx.x] += sred[threadIdx.x + st];
    __syncthreads();
  }
  if (threadIdx.x == 0) atomicAdd(lossAcc, sred[0]);
}

// ---------------- cos_t: row max then sum relu(max - cos - 0.55) ----------------
__global__ __launch_bounds__(256) void k_reduce_t(const float* __restrict__ cosT,
                                                  float* __restrict__ lossAcc)
{
  __shared__ float sred[256];
  const int row = blockIdx.x;  // grid = M_ROWS
  const float* rp = cosT + (size_t)row * NE_T;
  float mx = -3.4e38f;
  for (int c = threadIdx.x; c < NE_T; c += 256) mx = fmaxf(mx, rp[c]);
  sred[threadIdx.x] = mx; __syncthreads();
  for (int st = 128; st > 0; st >>= 1) {
    if (threadIdx.x < st) sred[threadIdx.x] = fmaxf(sred[threadIdx.x], sred[threadIdx.x + st]);
    __syncthreads();
  }
  const float rowmax = sred[0];
  __syncthreads();
  float s = 0.f;
  for (int c = threadIdx.x; c < NE_T; c += 256) {
    const float t = rowmax - rp[c] - 0.55f;
    s += t > 0.f ? t : 0.f;
  }
  sred[threadIdx.x] = s; __syncthreads();
  for (int st = 128; st > 0; st >>= 1) {
    if (threadIdx.x < st) sred[threadIdx.x] += sred[threadIdx.x + st];
    __syncthreads();
  }
  if (threadIdx.x == 0) atomicAdd(lossAcc, sred[0]);
}

// ---------------- scalar loss assembly ----------------
__global__ void k_finalize(const float* __restrict__ acc,
                           float* __restrict__ outCbl, float* __restrict__ outCos)
{
  if (threadIdx.x == 0 && blockIdx.x == 0) {
    // cbl = (1+BETA) * (sum_s + sum_t) / IMG_N
    *outCbl = 1.25f * acc[0] / (float)IMG_N;
    *outCos = acc[1] / ((float)M_ROWS * (float)NE_S)
            + acc[2] / ((float)M_ROWS * (float)NE_T);
  }
}

// ---------------- host orchestration ----------------
extern "C" void kernel_launch(void* const* d_in, const int* in_sizes, int n_in,
                              void* d_out, int out_size, void* d_ws, size_t ws_size,
                              hipStream_t stream) {
  (void)in_sizes; (void)n_in; (void)out_size; (void)ws_size;
  const float* z    = (const float*)d_in[0];
  const float* cb_s = (const float*)d_in[1];
  const float* cb_t = (const float*)d_in[2];

  float* out      = (float*)d_out;
  float* out_zq   = out;                            // IMG_N
  float* out_cbl  = out + IMG_N;                    // 1
  float* out_cos  = out + IMG_N + 1;                // 1
  float* out_idxs = out + IMG_N + 2;                // M_ROWS*6
  float* out_idxt = out_idxs + (size_t)M_ROWS * RVQ_DEPTH;

  char* w = (char*)d_ws;
  auto carve = [&](size_t bytes) -> void* {
    void* p = (void*)w;
    w += (bytes + 255) & ~(size_t)255;
    return p;
  };
  float*          r      = (float*)         carve((size_t)M_PAD * EDIM * 4);
  unsigned short* rbf    = (unsigned short*)carve((size_t)M_PAD * EDIM * 2);
  float*          G      = (float*)         carve((size_t)M_PAD * EDIM * 4); // zq_flat & cos_t
  unsigned short* cbsbf  = (unsigned short*)carve((size_t)NE_S * EDIM * 2);
  unsigned short* cbtbf  = (unsigned short*)carve((size_t)NE_T * EDIM * 2);
  float*          cbssq  = (float*)carve(NE_S * 4);
  float*          invcs  = (float*)carve(NE_S * 4);
  float*          cbtsq  = (float*)carve(NE_T * 4);
  float*          invct  = (float*)carve(NE_T * 4);
  float*          invRow = (float*)carve((size_t)M_PAD * 4);
  int*            idxT   = (int*)  carve((size_t)M_PAD * 4);
  float*          acc    = (float*)carve(16 * 4);   // [0]=cbl sum [1]=cos_s [2]=cos_t

  hipMemsetAsync(acc, 0, 16 * sizeof(float), stream);

  k_unfold<<<M_PAD, 256, 0, stream>>>(z, r, rbf, invRow);
  k_cbprep<<<NE_S, 256, 0, stream>>>(cb_s, cbsbf, cbssq, invcs);
  k_cbprep<<<NE_T, 256, 0, stream>>>(cb_t, cbtbf, cbtsq, invct);

  const int gemmGrid = M_PAD / 128;  // 249

  // cosine losses while rbf == bf16(zf)
  k_gemm<1><<<gemmGrid, 256, 0, stream>>>(rbf, cbsbf, invcs, invRow,
                                          nullptr, nullptr, acc + 1, NE_S);
  k_gemm<2><<<gemmGrid, 256, 0, stream>>>(rbf, cbtbf, invct, invRow,
                                          nullptr, G, nullptr, NE_T);
  k_reduce_t<<<M_ROWS, 256, 0, stream>>>(G, acc + 2);

  // shared-codebook residual VQ
  hipMemsetAsync(G, 0, (size_t)M_PAD * EDIM * 4, stream);
  for (int d = 0; d < RVQ_DEPTH; ++d) {
    k_gemm<0><<<gemmGrid, 256, 0, stream>>>(rbf, cbsbf, cbssq, nullptr,
                                            idxT, nullptr, nullptr, NE_S);
    k_update<<<M_ROWS, 128, 0, stream>>>(r, rbf, G, cb_s, idxT, out_idxs, d);
  }
  k_fold<<<(IMG_N + 255) / 256, 256, 0, stream>>>(G, z, out_zq, 0.7f, 1, acc + 0);

  // reset residual, task-codebook residual VQ
  k_unfold<<<M_PAD, 256, 0, stream>>>(z, r, rbf, invRow);
  hipMemsetAsync(G, 0, (size_t)M_PAD * EDIM * 4, stream);
  for (int d = 0; d < RVQ_DEPTH; ++d) {
    k_gemm<0><<<gemmGrid, 256, 0, stream>>>(rbf, cbtbf, cbtsq, nullptr,
                                            idxT, nullptr, nullptr, NE_T);
    k_update<<<M_ROWS, 128, 0, stream>>>(r, rbf, G, cb_t, idxT, out_idxt, d);
  }
  k_fold<<<(IMG_N + 255) / 256, 256, 0, stream>>>(G, z, out_zq, 0.3f, 0, acc + 0);

  k_finalize<<<1, 1, 0, stream>>>(acc, out_cbl, out_cos);
}
